// NGCWithPretrainedTKGAndTimestamps_86363202388691
// MI455X (gfx1250) — compile-verified, matrix operand order checked by reading, not run on hardware
//
#include <hip/hip_runtime.h>

#define D_TS   128
#define E_ENT  64
#define E_TSE  64
#define DIN    256
#define H1     4
#define C1     32
#define HC1    128
#define OUTC   32
#define HID    32
#define LAT    16
#define NEG_SLOPE 0.2f
#define REG_W  0.01f

typedef __attribute__((ext_vector_type(16))) _Float16 v16h;
typedef __attribute__((ext_vector_type(8)))  float    v8f;

// ---------------------------------------------------------------------------
// WMMA GEMM: C[M,NC] = A[M,K] @ B[K,NC] + bias[NC], optional ReLU.
// K, NC compile-time (K multiple of 16; NC multiple of 16). One wave per
// 16x16 output tile; K stepped by 32, fully unrolled; the upper 8-element
// chunk of the last fragment is statically zeroed when K%32==16.
// A fragment: lane holds row tm*16+(lane&15); lanes 0-15 cover K=k0+{0..7,
// 16..23}, lanes 16-31 cover K=k0+{8..15,24..31}. B mirrors with lane=col.
// ---------------------------------------------------------------------------
template <int K, int NC>
__global__ void wmma_gemm_kernel(const float* __restrict__ A,
                                 const float* __restrict__ B,
                                 const float* __restrict__ bias,
                                 float* __restrict__ C,
                                 int M, int act)
{
    constexpr int tilesN = NC / 16;
    const int wave = threadIdx.x >> 5;
    const int lane = threadIdx.x & 31;
    const int tilesM = M >> 4;
    const int tile = blockIdx.x * (blockDim.x >> 5) + wave;
    if (tile >= tilesM * tilesN) return;          // wave-uniform: EXEC stays all-1s
    const int tm = tile / tilesN;
    const int tn = tile % tilesN;

    const int sub  = lane & 15;          // row (A) / col (B) within tile
    const int kgrp = (lane >> 4) << 3;   // 0 or 8
    const long row = (long)tm * 16 + sub;
    const int  col = tn * 16 + sub;

    const float* __restrict__ Al = A + row * (long)K + kgrp;     // lane base in A
    const float* __restrict__ Bl = B + (long)kgrp * NC + col;    // lane base in B

    v8f acc = {};
#pragma unroll
    for (int k0 = 0; k0 < K; k0 += 32) {
        v16h af, bf;
        // chunk 0: K = k0 + kgrp + {0..7}  (always valid: k0+16 <= K)
        {
            float4 f0 = *(const float4*)(Al + k0);
            float4 f1 = *(const float4*)(Al + k0 + 4);
            af[0] = (_Float16)f0.x; af[1] = (_Float16)f0.y;
            af[2] = (_Float16)f0.z; af[3] = (_Float16)f0.w;
            af[4] = (_Float16)f1.x; af[5] = (_Float16)f1.y;
            af[6] = (_Float16)f1.z; af[7] = (_Float16)f1.w;
#pragma unroll
            for (int i = 0; i < 8; ++i)
                bf[i] = (_Float16)Bl[(long)(k0 + i) * NC];
        }
        // chunk 1: K = k0 + kgrp + 16 + {0..7}  (valid iff k0+32 <= K; constant)
        if (k0 + 32 <= K) {
            float4 f0 = *(const float4*)(Al + k0 + 16);
            float4 f1 = *(const float4*)(Al + k0 + 20);
            af[8]  = (_Float16)f0.x; af[9]  = (_Float16)f0.y;
            af[10] = (_Float16)f0.z; af[11] = (_Float16)f0.w;
            af[12] = (_Float16)f1.x; af[13] = (_Float16)f1.y;
            af[14] = (_Float16)f1.z; af[15] = (_Float16)f1.w;
#pragma unroll
            for (int i = 0; i < 8; ++i)
                bf[8 + i] = (_Float16)Bl[(long)(k0 + 16 + i) * NC];
        } else {
#pragma unroll
            for (int i = 0; i < 8; ++i) {
                af[8 + i] = (_Float16)0.0f;
                bf[8 + i] = (_Float16)0.0f;
            }
        }
        acc = __builtin_amdgcn_wmma_f32_16x16x32_f16(
            false, af, false, bf, (short)0, acc, false, false);
    }

    // D layout: VGPR r -> M = tm*16 + r (+8 for lanes 16-31), N = lane&15
    const long rbase = (long)tm * 16 + ((lane >> 4) ? 8 : 0);
    const float bv = bias ? bias[col] : 0.0f;
#pragma unroll
    for (int r = 0; r < 8; ++r) {
        float v = acc[r] + bv;
        if (act) v = v > 0.0f ? v : 0.0f;
        C[(rbase + r) * (long)NC + col] = v;
    }
}

// ---------------------------------------------------------------------------
// Elementwise / gather kernels
// ---------------------------------------------------------------------------
__global__ void gather_concat_kernel(const float* __restrict__ ts,
                                     const float* __restrict__ etab,
                                     const float* __restrict__ ttab,
                                     const int* __restrict__ eidx,
                                     const int* __restrict__ tidx,
                                     float* __restrict__ x, int n)
{
    long i = (long)blockIdx.x * blockDim.x + threadIdx.x;
    long total = (long)n * DIN;
    if (i >= total) return;
    int node = (int)(i / DIN);
    int j = (int)(i % DIN);
    float v;
    if (j < D_TS)              v = ts[(long)node * D_TS + j];
    else if (j < D_TS + E_ENT) v = etab[(long)eidx[node] * E_ENT + (j - D_TS)];
    else                       v = ttab[(long)tidx[node] * E_TSE + (j - D_TS - E_ENT)];
    x[i] = v;
}

__global__ void zero_kernel(float* __restrict__ p, long n)
{
    long i = (long)blockIdx.x * blockDim.x + threadIdx.x;
    if (i < n) p[i] = 0.0f;
}

__global__ void bias_act_kernel(const float* __restrict__ in, const float* __restrict__ b,
                                float* __restrict__ out, long n, int ncol, int relu)
{
    long i = (long)blockIdx.x * blockDim.x + threadIdx.x;
    if (i >= n) return;
    float v = in[i] + b[(int)(i % ncol)];
    if (relu) v = v > 0.0f ? v : 0.0f;
    out[i] = v;
}

// ---------------------------------------------------------------------------
// Edge (segment) kernels. Self loops appended after the raw edges.
// ---------------------------------------------------------------------------
__device__ __forceinline__ void edge_sd(const int* __restrict__ ei, int e, int eraw,
                                        int& s, int& d)
{
    if (e < eraw) { s = ei[e]; d = ei[eraw + e]; }
    else          { s = e - eraw; d = s; }
}

// score = dot(leaky_relu(xl[src]+xr[dst]), att[h]);  segment-max via
// monotone uint keys + atomicMax.
__global__ void edge_score_kernel(const float* __restrict__ xl, const float* __restrict__ xr,
                                  const float* __restrict__ att, const int* __restrict__ ei,
                                  float* __restrict__ scores, unsigned* __restrict__ emaxk,
                                  int eraw, int etot, int heads, int ch)
{
    long t = (long)blockIdx.x * blockDim.x + threadIdx.x;
    if (t >= (long)etot * heads) return;
    int e = (int)(t / heads), h = (int)(t % heads);
    int s, d; edge_sd(ei, e, eraw, s, d);
    const int hc = heads * ch;
    const float4* pl = (const float4*)(xl + (long)s * hc + h * ch);
    const float4* pr = (const float4*)(xr + (long)d * hc + h * ch);
    const float4* pa = (const float4*)(att + h * ch);
    float sc = 0.0f;
    for (int c = 0; c < (ch >> 2); ++c) {
        float4 a = pl[c], b = pr[c], w = pa[c];
        float m;
        m = a.x + b.x; m = m > 0.0f ? m : NEG_SLOPE * m; sc += m * w.x;
        m = a.y + b.y; m = m > 0.0f ? m : NEG_SLOPE * m; sc += m * w.y;
        m = a.z + b.z; m = m > 0.0f ? m : NEG_SLOPE * m; sc += m * w.z;
        m = a.w + b.w; m = m > 0.0f ? m : NEG_SLOPE * m; sc += m * w.w;
    }
    scores[t] = sc;
    unsigned u = __float_as_uint(sc);
    unsigned key = u ^ ((unsigned)(((int)u) >> 31) | 0x80000000u);
    atomicMax(emaxk + (long)d * heads + h, key);
}

// a = exp(score - segmax); in-place over scores; segment-sum into denom.
__global__ void edge_exp_kernel(float* __restrict__ scores, const unsigned* __restrict__ emaxk,
                                const int* __restrict__ ei, float* __restrict__ denom,
                                int eraw, int etot, int heads)
{
    long t = (long)blockIdx.x * blockDim.x + threadIdx.x;
    if (t >= (long)etot * heads) return;
    int e = (int)(t / heads), h = (int)(t % heads);
    int s, d; edge_sd(ei, e, eraw, s, d);
    unsigned key = emaxk[(long)d * heads + h];
    unsigned u = (key & 0x80000000u) ? (key ^ 0x80000000u) : ~key;
    float mx = __uint_as_float(u);
    float a = expf(scores[t] - mx);
    scores[t] = a;
    atomicAdd(denom + (long)d * heads + h, a);
}

// out[dst] += (a/denom[dst]) * xl[src]
__global__ void edge_aggregate_kernel(const float* __restrict__ xl, const float* __restrict__ aexp,
                                      const float* __restrict__ denom, const int* __restrict__ ei,
                                      float* __restrict__ agg, int eraw, int etot,
                                      int heads, int ch)
{
    long t = (long)blockIdx.x * blockDim.x + threadIdx.x;
    if (t >= (long)etot * heads) return;
    int e = (int)(t / heads), h = (int)(t % heads);
    int s, d; edge_sd(ei, e, eraw, s, d);
    float alpha = aexp[t] / denom[(long)d * heads + h];
    const int hc = heads * ch;
    const float* pl = xl + (long)s * hc + h * ch;
    float* po = agg + (long)d * hc + h * ch;
    for (int c = 0; c < ch; ++c) atomicAdd(po + c, alpha * pl[c]);
}

// ---------------------------------------------------------------------------
// VAE reparameterization + regularization reduction
// ---------------------------------------------------------------------------
__global__ void vae_z_kernel(const float* __restrict__ q, const float* __restrict__ eps,
                             float* __restrict__ zout, float* __restrict__ mout,
                             float* __restrict__ lvout, long n)
{
    long i = (long)blockIdx.x * blockDim.x + threadIdx.x;
    if (i >= n) return;
    long node = i >> 4;
    int l = (int)(i & 15);
    float mean = q[node * 32 + l];
    float lv   = q[node * 32 + 16 + l];
    float z    = mean + expf(0.5f * lv) * eps[i];
    mout[i] = mean; lvout[i] = lv; zout[i] = z;
}

__global__ void reg_reduce_kernel(const float* __restrict__ v, float* __restrict__ out, long n)
{
    __shared__ float sm[256];
    long i = (long)blockIdx.x * blockDim.x + threadIdx.x;
    float p = 0.0f;
    if (i < n) { float q = v[i]; p = q * q; }
    sm[threadIdx.x] = p;
    __syncthreads();
    for (int s = 128; s > 0; s >>= 1) {
        if ((int)threadIdx.x < s) sm[threadIdx.x] += sm[threadIdx.x + s];
        __syncthreads();
    }
    if (threadIdx.x == 0) atomicAdd(out, REG_W * sm[0]);
}

// ---------------------------------------------------------------------------
// Host launch
// ---------------------------------------------------------------------------
static inline int nblk(long total, int bs) { return (int)((total + bs - 1) / bs); }

extern "C" void kernel_launch(void* const* d_in, const int* in_sizes, int n_in,
                              void* d_out, int out_size, void* d_ws, size_t ws_size,
                              hipStream_t stream)
{
    const float* ts    = (const float*)d_in[0];
    const float* eps   = (const float*)d_in[1];
    const float* etab  = (const float*)d_in[2];
    const float* ttab  = (const float*)d_in[3];
    const float* Wl1   = (const float*)d_in[4];
    const float* bl1   = (const float*)d_in[5];
    const float* Wr1   = (const float*)d_in[6];
    const float* br1   = (const float*)d_in[7];
    const float* att1  = (const float*)d_in[8];
    const float* bias1 = (const float*)d_in[9];
    const float* Wl2   = (const float*)d_in[10];
    const float* bl2   = (const float*)d_in[11];
    const float* Wr2   = (const float*)d_in[12];
    const float* br2   = (const float*)d_in[13];
    const float* att2  = (const float*)d_in[14];
    const float* bias2 = (const float*)d_in[15];
    const float* We1   = (const float*)d_in[16];
    const float* be1   = (const float*)d_in[17];
    const float* We2   = (const float*)d_in[18];
    const float* be2   = (const float*)d_in[19];
    const float* Wd1   = (const float*)d_in[20];
    const float* bd1   = (const float*)d_in[21];
    const float* Wd2   = (const float*)d_in[22];
    const float* bd2   = (const float*)d_in[23];
    const int* eidx    = (const int*)d_in[24];
    // d_in[25] = relation_indices (unused by reference)
    const int* tidx    = (const int*)d_in[26];
    const int* edge    = (const int*)d_in[27];

    const int n    = in_sizes[24];          // 50000
    const int eraw = in_sizes[27] / 2;      // 800000
    const int etot = eraw + n;              // + self loops

    // ---- workspace layout (floats), with reuse ----
    float* w      = (float*)d_ws;
    float* x      = w;                               // [n,256] (reused as agg)
    float* xl     = x  + (long)n * DIN;              // [n,128] (layer2: [n,32]; VAE: v1)
    float* xr     = xl + (long)n * HC1;              // [n,128] (layer2: [n,32]; VAE: v2)
    float* hb     = xr + (long)n * HC1;              // [n,128] relu(gat1)   (VAE: v3)
    float* scores = hb + (long)n * HC1;              // [etot,4] (in-place exp)
    unsigned* emaxk = (unsigned*)(scores + (long)etot * H1);   // [n,4]
    float* denom  = (float*)(emaxk + (long)n * H1);  // [n,4]
    float* agg    = x;
    float* v1 = xl, *v2 = xr, *v3 = hb;

    // ---- output layout ----
    float* out    = (float*)d_out;
    float* o_caus = out;
    float* o_z    = o_caus + (long)n * OUTC;
    float* o_mean = o_z    + (long)n * LAT;
    float* o_lv   = o_mean + (long)n * LAT;
    float* o_xrec = o_lv   + (long)n * LAT;
    float* o_reg  = o_xrec + (long)n * D_TS;

    // 1. x = concat(ts, entity_emb, timestamp_emb)
    gather_concat_kernel<<<nblk((long)n * DIN, 256), 256, 0, stream>>>(
        ts, etab, ttab, eidx, tidx, x, n);

    // 2. GAT1 linear transforms (WMMA, K=256, NC=128)
    {
        int tiles = (n / 16) * (HC1 / 16);
        wmma_gemm_kernel<DIN, HC1><<<nblk(tiles, 8), 256, 0, stream>>>(x, Wl1, bl1, xl, n, 0);
        wmma_gemm_kernel<DIN, HC1><<<nblk(tiles, 8), 256, 0, stream>>>(x, Wr1, br1, xr, n, 0);
    }

    // 3. layer-1 segment softmax + aggregation
    zero_kernel<<<nblk((long)n * HC1, 256), 256, 0, stream>>>(agg, (long)n * HC1);
    zero_kernel<<<nblk((long)n * H1, 256), 256, 0, stream>>>((float*)emaxk, (long)n * H1);
    zero_kernel<<<nblk((long)n * H1, 256), 256, 0, stream>>>(denom, (long)n * H1);
    const long eh1 = (long)etot * H1;
    edge_score_kernel<<<nblk(eh1, 256), 256, 0, stream>>>(
        xl, xr, att1, edge, scores, emaxk, eraw, etot, H1, C1);
    edge_exp_kernel<<<nblk(eh1, 256), 256, 0, stream>>>(
        scores, emaxk, edge, denom, eraw, etot, H1);
    edge_aggregate_kernel<<<nblk(eh1, 256), 256, 0, stream>>>(
        xl, scores, denom, edge, agg, eraw, etot, H1, C1);
    bias_act_kernel<<<nblk((long)n * HC1, 256), 256, 0, stream>>>(
        agg, bias1, hb, (long)n * HC1, HC1, 1);

    // 4. GAT2 linear transforms (WMMA, K=128, NC=32)
    {
        int tiles = (n / 16) * (OUTC / 16);
        wmma_gemm_kernel<HC1, OUTC><<<nblk(tiles, 8), 256, 0, stream>>>(hb, Wl2, bl2, xl, n, 0);
        wmma_gemm_kernel<HC1, OUTC><<<nblk(tiles, 8), 256, 0, stream>>>(hb, Wr2, br2, xr, n, 0);
    }

    // 5. layer-2 segment softmax + aggregation (H=1, C=32)
    zero_kernel<<<nblk((long)n * OUTC, 256), 256, 0, stream>>>(agg, (long)n * OUTC);
    zero_kernel<<<nblk((long)n, 256), 256, 0, stream>>>((float*)emaxk, (long)n);
    zero_kernel<<<nblk((long)n, 256), 256, 0, stream>>>(denom, (long)n);
    const long eh2 = (long)etot;
    edge_score_kernel<<<nblk(eh2, 256), 256, 0, stream>>>(
        xl, xr, att2, edge, scores, emaxk, eraw, etot, 1, OUTC);
    edge_exp_kernel<<<nblk(eh2, 256), 256, 0, stream>>>(
        scores, emaxk, edge, denom, eraw, etot, 1);
    edge_aggregate_kernel<<<nblk(eh2, 256), 256, 0, stream>>>(
        xl, scores, denom, edge, agg, eraw, etot, 1, OUTC);
    bias_act_kernel<<<nblk((long)n * OUTC, 256), 256, 0, stream>>>(
        agg, bias2, o_caus, (long)n * OUTC, OUTC, 0);

    // 6. regularization = 0.01 * sum(caus^2)
    zero_kernel<<<1, 64, 0, stream>>>(o_reg, 1);
    reg_reduce_kernel<<<nblk((long)n * OUTC, 256), 256, 0, stream>>>(
        o_caus, o_reg, (long)n * OUTC);

    // 7. VAE (WMMA; decoder K=16 statically zero-padded)
    {
        int tiles = (n / 16) * (HID / 16);
        wmma_gemm_kernel<D_TS, HID><<<nblk(tiles, 8), 256, 0, stream>>>(ts, We1, be1, v1, n, 1);
        tiles = (n / 16) * ((2 * LAT) / 16);
        wmma_gemm_kernel<HID, 2 * LAT><<<nblk(tiles, 8), 256, 0, stream>>>(v1, We2, be2, v2, n, 0);
    }
    vae_z_kernel<<<nblk((long)n * LAT, 256), 256, 0, stream>>>(
        v2, eps, o_z, o_mean, o_lv, (long)n * LAT);
    {
        int tiles = (n / 16) * (HID / 16);
        wmma_gemm_kernel<LAT, HID><<<nblk(tiles, 8), 256, 0, stream>>>(o_z, Wd1, bd1, v3, n, 1);
        tiles = (n / 16) * (D_TS / 16);
        wmma_gemm_kernel<HID, D_TS><<<nblk(tiles, 8), 256, 0, stream>>>(v3, Wd2, bd2, o_xrec, n, 0);
    }
}
